// QuantAhpcNetwork_21071109554523
// MI455X (gfx1250) — 16-bit and 8-bit matrix operand order repaired and checked by reading, compiled, not run on hardware
//
#include <hip/hip_runtime.h>
#include <stdint.h>
#include <stddef.h>

typedef __attribute__((ext_vector_type(16))) __bf16       v16bf;
typedef __attribute__((ext_vector_type(8)))  float        v8f;
typedef __attribute__((ext_vector_type(8)))  int          v8i;
typedef __attribute__((ext_vector_type(4)))  unsigned int u32x4;
typedef __attribute__((ext_vector_type(2)))  unsigned int u32x2;
typedef __attribute__((ext_vector_type(4)))  float        f32x4;
typedef __attribute__((ext_vector_type(8)))  int          i32x8;
typedef __attribute__((ext_vector_type(4)))  int          i32x4;

#define B_ALL   256
#define T_STEPS 100
#define F_DIM   1024
#define H1      512
#define H2      512
#define O_DIM   10

#define BT      16
#define THREADS 512
#define WAVES   16
#define TPW     2

#define SMEM_STAGE  (BT * F_DIM * 4)
#define SMEM_BYTES  (2*SMEM_STAGE + BT*F_DIM*2*2 + 3*BT*H2)

__device__ __forceinline__ v8f wmma_bf16(v16bf a, v16bf b, v8f c) {
  return __builtin_amdgcn_wmma_f32_16x16x32_bf16(false, a, false, b, (short)0, c, false, false);
}
__device__ __forceinline__ v8i wmma_iu8(v8i a, v8i b, v8i c) {
  return __builtin_amdgcn_wmma_i32_16x16x64_iu8(false, a, true, b, c, false, false);
}

__device__ __forceinline__ void tdm_load_2d_f32(unsigned lds_off, const void* gptr,
                                                unsigned rowElems, unsigned rows,
                                                unsigned long long rowStrideElems) {
  unsigned long long ga = (unsigned long long)(uintptr_t)gptr;
  u32x4 g0;
  g0[0] = 1u;
  g0[1] = lds_off;
  g0[2] = (unsigned)(ga & 0xffffffffu);
  g0[3] = (unsigned)((ga >> 32) & 0x01ffffffu)
        | (2u << 30);
  i32x8 g1;
  g1[0] = (int)(2u << 16);
  g1[1] = (int)((rowElems & 0xffffu) << 16);
  g1[2] = (int)(((rowElems >> 16) & 0xffffu)
        | ((rows & 0xffffu) << 16));
  g1[3] = (int)(((rows >> 16) & 0xffffu)
        | ((rowElems & 0xffffu) << 16));
  g1[4] = (int)(rows & 0xffffu);
  g1[5] = (int)(rowStrideElems & 0xffffffffu);
  g1[6] = (int)((rowStrideElems >> 32) & 0xffffu);
  g1[7] = 0;
  i32x4 z4 = {0, 0, 0, 0};
  i32x8 z8 = {0, 0, 0, 0, 0, 0, 0, 0};
  __builtin_amdgcn_tensor_load_to_lds(g0, g1, z4, z4, z8, 0);
}

__device__ __forceinline__ v16bf load_a16(const __bf16* base, int lane, int kbase, int ldk) {
  int m  = lane & 15;
  int k0 = kbase + ((lane >> 4) << 3);
  const __bf16* p = base + m * ldk + k0;
  v16bf r;
  ((u32x4*)&r)[0] = *(const u32x4*)(p);
  ((u32x4*)&r)[1] = *(const u32x4*)(p + 16);
  return r;
}
__device__ __forceinline__ v16bf load_b16(const __bf16* w, int lane, int n0, int kbase, int ldk) {
  int n  = n0 + (lane & 15);
  int k0 = kbase + ((lane >> 4) << 3);
  const __bf16* p = w + (size_t)n * ldk + k0;
  v16bf r;
  ((u32x4*)&r)[0] = *(const u32x4*)(p);
  ((u32x4*)&r)[1] = *(const u32x4*)(p + 16);
  return r;
}
__device__ __forceinline__ v8i load_a8(const unsigned char* s, int lane, int kbase, int ldk) {
  int m = lane & 15;
  const unsigned char* p = s + m * ldk + kbase + ((lane >> 4) << 3);
  v8i r;
  ((uint64_t*)&r)[0] = *(const uint64_t*)(p);
  ((uint64_t*)&r)[1] = *(const uint64_t*)(p + 16);
  ((uint64_t*)&r)[2] = *(const uint64_t*)(p + 32);
  ((uint64_t*)&r)[3] = *(const uint64_t*)(p + 48);
  return r;
}
__device__ __forceinline__ v8i load_b8(const signed char* w, int lane, int n0, int kbase, int ldk) {
  int n = n0 + (lane & 15);
  const signed char* p = w + (size_t)n * ldk + kbase + ((lane >> 4) << 3);
  v8i r;
  ((uint64_t*)&r)[0] = *(const uint64_t*)(p);
  ((uint64_t*)&r)[1] = *(const uint64_t*)(p + 16);
  ((uint64_t*)&r)[2] = *(const uint64_t*)(p + 32);
  ((uint64_t*)&r)[3] = *(const uint64_t*)(p + 48);
  return r;
}

__global__ void k_absmax(const float* __restrict__ w, int n, unsigned int* slot) {
  float m = 0.f;
  for (int i = blockIdx.x * blockDim.x + threadIdx.x; i < n; i += gridDim.x * blockDim.x)
    m = fmaxf(m, fabsf(w[i]));
  atomicMax(slot, __float_as_uint(m));
}

__global__ void k_quant_bf16(const float* __restrict__ w, int n,
                             const unsigned int* __restrict__ bits,
                             float* __restrict__ scaleOut, __bf16* __restrict__ codes) {
  float amax = __uint_as_float(*bits);
  float s    = amax / 127.f;
  float inv  = (s > 0.f) ? (1.f / s) : 0.f;
  if (blockIdx.x == 0 && threadIdx.x == 0) *scaleOut = s;
  for (int i = blockIdx.x * blockDim.x + threadIdx.x; i < n; i += gridDim.x * blockDim.x) {
    float q = rintf(w[i] * inv);
    q = fminf(127.f, fmaxf(-128.f, q));
    codes[i] = (__bf16)q;
  }
}

__global__ void k_quant_i8(const float* __restrict__ w, int rows_valid, int rows_total, int cols,
                           const unsigned int* __restrict__ bits,
                           float* __restrict__ scaleOut, signed char* __restrict__ codes) {
  float amax = __uint_as_float(*bits);
  float s    = amax / 127.f;
  float inv  = (s > 0.f) ? (1.f / s) : 0.f;
  if (blockIdx.x == 0 && threadIdx.x == 0) *scaleOut = s;
  int n = rows_total * cols;
  for (int i = blockIdx.x * blockDim.x + threadIdx.x; i < n; i += gridDim.x * blockDim.x) {
    int r = i / cols, c = i - r * cols;
    float q = 0.f;
    if (r < rows_valid) {
      q = rintf(w[r * cols + c] * inv);
      q = fminf(127.f, fmaxf(-128.f, q));
    }
    codes[i] = (signed char)q;
  }
}

__global__ void __launch_bounds__(THREADS, 1)
snn_step_kernel(const float* __restrict__ data,
                const __bf16* __restrict__ W1c,
                const signed char* __restrict__ W2c,
                const signed char* __restrict__ Winc,
                const signed char* __restrict__ Woutc,
                const signed char* __restrict__ W3c,
                const float* __restrict__ scales,
                const float* __restrict__ beta_in, const float* __restrict__ beta_rec,
                const float* __restrict__ beta_back, const float* __restrict__ beta_out,
                const float* __restrict__ vth_in, const float* __restrict__ vth_rec,
                const float* __restrict__ vth_back, const float* __restrict__ vth_out,
                float* __restrict__ out) {
  extern __shared__ unsigned char smem[];
  float*  xstg[2];
  xstg[0]            = (float*)smem;
  xstg[1]            = xstg[0] + BT * F_DIM;
  __bf16* xhi        = (__bf16*)(xstg[1] + BT * F_DIM);
  __bf16* xlo        = xhi + BT * F_DIM;
  unsigned char* s1m = (unsigned char*)(xlo + BT * F_DIM);
  unsigned char* rsm = s1m + BT * H1;
  unsigned char* bsm = rsm + BT * H2;

  const int tid  = threadIdx.x;
  const int lane = tid & 31;
  const int wave = tid >> 5;
  const int b0   = blockIdx.x * BT;
  const int nt0  = wave * (TPW * 16);
  const int mrow = (lane >> 4) << 3;
  const int ncol = lane & 15;

  const float sc1 = scales[0], sc2 = scales[1], sci = scales[2],
              sco = scales[3], sc3 = scales[4];
  const float b1 = fminf(1.f, fmaxf(0.f, beta_in[0]));
  const float br = fminf(1.f, fmaxf(0.f, beta_rec[0]));
  const float bb = fminf(1.f, fmaxf(0.f, beta_back[0]));
  const float bo = fminf(1.f, fmaxf(0.f, beta_out[0]));
  const float vin = vth_in[0], vrec = vth_rec[0], vback = vth_back[0], vout = vth_out[0];

  v8f m1s[TPW], rm[TPW], bm[TPW], m2;
  #pragma unroll
  for (int j = 0; j < TPW; ++j)
    #pragma unroll
    for (int vv = 0; vv < 8; ++vv) { m1s[j][vv] = 0.f; rm[j][vv] = 0.f; bm[j][vv] = 0.f; }
  #pragma unroll
  for (int vv = 0; vv < 8; ++vv) m2[vv] = 0.f;

  for (int i = tid; i < BT * H2; i += THREADS) rsm[i] = 0;

  if (wave == 0) {
    tdm_load_2d_f32((unsigned)(uintptr_t)xstg[0],
                    data + (size_t)b0 * T_STEPS * F_DIM,
                    F_DIM, BT, (unsigned long long)T_STEPS * F_DIM);
  }
  __syncthreads();

  #pragma unroll 1
  for (int t = 0; t < T_STEPS; ++t) {
    const int buf = t & 1;
    if (wave == 0) __builtin_amdgcn_s_wait_tensorcnt(0);
    __syncthreads();
    if (wave == 0 && (t + 1) < T_STEPS) {
      tdm_load_2d_f32((unsigned)(uintptr_t)xstg[buf ^ 1],
                      data + ((size_t)b0 * T_STEPS + (t + 1)) * F_DIM,
                      F_DIM, BT, (unsigned long long)T_STEPS * F_DIM);
    }

    for (int e = tid; e < BT * (F_DIM / 4); e += THREADS) {
      int r  = e >> 8;
      int c4 = e & 255;
      f32x4 v = *(const f32x4*)(xstg[buf] + r * F_DIM + (c4 << 2));
      union { __bf16 h[4]; u32x2 u; } ph, pl;
      #pragma unroll
      for (int q = 0; q < 4; ++q) {
        float x = v[q];
        __bf16 h = (__bf16)x;
        ph.h[q] = h;
        pl.h[q] = (__bf16)(x - (float)h);
      }
      *(u32x2*)(xhi + r * F_DIM + (c4 << 2)) = ph.u;
      *(u32x2*)(xlo + r * F_DIM + (c4 << 2)) = pl.u;
    }
    __syncthreads();

    {
      v8f acc[TPW];
      #pragma unroll
      for (int j = 0; j < TPW; ++j)
        #pragma unroll
        for (int vv = 0; vv < 8; ++vv) acc[j][vv] = 0.f;
      #pragma unroll 2
      for (int kt = 0; kt < F_DIM / 32; ++kt) {
        v16bf ah = load_a16(xhi, lane, kt * 32, F_DIM);
        v16bf al = load_a16(xlo, lane, kt * 32, F_DIM);
        #pragma unroll
        for (int j = 0; j < TPW; ++j) {
          v16bf bf = load_b16(W1c, lane, nt0 + j * 16, kt * 32, F_DIM);
          acc[j] = wmma_bf16(ah, bf, acc[j]);
          acc[j] = wmma_bf16(al, bf, acc[j]);
        }
      }
      #pragma unroll
      for (int j = 0; j < TPW; ++j)
        #pragma unroll
        for (int vv = 0; vv < 8; ++vv) {
          float mm = b1 * m1s[j][vv] + sc1 * acc[j][vv];
          float sp = (mm - vin) > 0.f ? 1.f : 0.f;
          m1s[j][vv] = mm * (1.f - sp);
          s1m[(mrow + vv) * H1 + nt0 + j * 16 + ncol] = (unsigned char)sp;
        }
    }
    __syncthreads();

    {
      v8i ia[TPW];
      #pragma unroll
      for (int j = 0; j < TPW; ++j) ia[j] = (v8i){0,0,0,0,0,0,0,0};
      #pragma unroll
      for (int kt = 0; kt < H1 / 64; ++kt) {
        v8i a = load_a8(s1m, lane, kt * 64, H1);
        #pragma unroll
        for (int j = 0; j < TPW; ++j)
          ia[j] = wmma_iu8(a, load_b8(W2c, lane, nt0 + j * 16, kt * 64, H1), ia[j]);
      }
      #pragma unroll
      for (int j = 0; j < TPW; ++j)
        #pragma unroll
        for (int vv = 0; vv < 8; ++vv)
          rm[j][vv] = br * rm[j][vv] + sc2 * (float)ia[j][vv];

      #pragma unroll
      for (int j = 0; j < TPW; ++j) ia[j] = (v8i){0,0,0,0,0,0,0,0};
      #pragma unroll
      for (int kt = 0; kt < H2 / 64; ++kt) {
        v8i a = load_a8(rsm, lane, kt * 64, H2);
        #pragma unroll
        for (int j = 0; j < TPW; ++j)
          ia[j] = wmma_iu8(a, load_b8(Winc, lane, nt0 + j * 16, kt * 64, H2), ia[j]);
      }
      #pragma unroll
      for (int j = 0; j < TPW; ++j)
        #pragma unroll
        for (int vv = 0; vv < 8; ++vv) {
          float mm = bb * bm[j][vv] + sci * (float)ia[j][vv];
          float sp = (mm - vback) > 0.f ? 1.f : 0.f;
          bm[j][vv] = mm * (1.f - sp);
          bsm[(mrow + vv) * H2 + nt0 + j * 16 + ncol] = (unsigned char)sp;
        }
    }
    __syncthreads();

    {
      v8i ia[TPW];
      #pragma unroll
      for (int j = 0; j < TPW; ++j) ia[j] = (v8i){0,0,0,0,0,0,0,0};
      #pragma unroll
      for (int kt = 0; kt < H2 / 64; ++kt) {
        v8i a = load_a8(bsm, lane, kt * 64, H2);
        #pragma unroll
        for (int j = 0; j < TPW; ++j)
          ia[j] = wmma_iu8(a, load_b8(Woutc, lane, nt0 + j * 16, kt * 64, H2), ia[j]);
      }
      #pragma unroll
      for (int j = 0; j < TPW; ++j)
        #pragma unroll
        for (int vv = 0; vv < 8; ++vv) {
          float mm = rm[j][vv] + sco * (float)ia[j][vv];
          float sp = (mm - vrec) > 0.f ? 1.f : 0.f;
          rm[j][vv] = mm * (1.f - sp);
          rsm[(mrow + vv) * H2 + nt0 + j * 16 + ncol] = (unsigned char)sp;
        }
    }
    __syncthreads();

    if (wave == 0) {
      v8i acc3 = (v8i){0,0,0,0,0,0,0,0};
      #pragma unroll
      for (int kt = 0; kt < H2 / 64; ++kt)
        acc3 = wmma_iu8(load_a8(rsm, lane, kt * 64, H2),
                        load_b8(W3c, lane, 0, kt * 64, H2), acc3);
      #pragma unroll
      for (int vv = 0; vv < 8; ++vv) {
        float mm = bo * m2[vv] + sc3 * (float)acc3[vv];
        float sp = (mm - vout) > 0.f ? 1.f : 0.f;
        m2[vv] = mm * (1.f - sp);
        if (ncol < O_DIM)
          out[(size_t)t * (B_ALL * O_DIM) + (size_t)(b0 + mrow + vv) * O_DIM + ncol] = sp;
      }
    }
  }
}

extern "C" void kernel_launch(void* const* d_in, const int* in_sizes, int n_in,
                              void* d_out, int out_size, void* d_ws, size_t ws_size,
                              hipStream_t stream) {
  const float* data = (const float*)d_in[0];
  const float* W1   = (const float*)d_in[1];
  const float* W2   = (const float*)d_in[2];
  const float* Win  = (const float*)d_in[3];
  const float* Wout = (const float*)d_in[4];
  const float* W3   = (const float*)d_in[5];

  unsigned int* absbits = (unsigned int*)d_ws;
  float*        scales  = (float*)((char*)d_ws + 32);
  __bf16*       W1c     = (__bf16*)((char*)d_ws + 64);
  signed char*  W2c     = (signed char*)((char*)d_ws + 64 + 512*1024*2);
  signed char*  Winc    = W2c  + 512*512;
  signed char*  Woutc   = Winc + 512*512;
  signed char*  W3c     = Woutc + 512*512;

  (void)hipMemsetAsync(d_ws, 0, 64, stream);

  k_absmax<<<256, 256, 0, stream>>>(W1,   512*1024, absbits + 0);
  k_absmax<<<128, 256, 0, stream>>>(W2,   512*512,  absbits + 1);
  k_absmax<<<128, 256, 0, stream>>>(Win,  512*512,  absbits + 2);
  k_absmax<<<128, 256, 0, stream>>>(Wout, 512*512,  absbits + 3);
  k_absmax<<<32,  256, 0, stream>>>(W3,   10*512,   absbits + 4);

  k_quant_bf16<<<512, 256, 0, stream>>>(W1, 512*1024, absbits + 0, scales + 0, W1c);
  k_quant_i8<<<256, 256, 0, stream>>>(W2,   512, 512, 512, absbits + 1, scales + 1, W2c);
  k_quant_i8<<<256, 256, 0, stream>>>(Win,  512, 512, 512, absbits + 2, scales + 2, Winc);
  k_quant_i8<<<256, 256, 0, stream>>>(Wout, 512, 512, 512, absbits + 3, scales + 3, Woutc);
  k_quant_i8<<<16,  256, 0, stream>>>(W3,   10,  16,  512, absbits + 4, scales + 4, W3c);

  snn_step_kernel<<<dim3(B_ALL / BT), dim3(THREADS), SMEM_BYTES, stream>>>(
      data, W1c, W2c, Winc, Woutc, W3c, scales,
      (const float*)d_in[6], (const float*)d_in[7], (const float*)d_in[8], (const float*)d_in[9],
      (const float*)d_in[10], (const float*)d_in[11], (const float*)d_in[12], (const float*)d_in[13],
      (float*)d_out);
}
